// HierarchicalPolicy_43971875176952
// MI455X (gfx1250) — matrix operand order repaired and checked by reading, not compiled, not run on hardware
//
#include <hip/hip_runtime.h>
#include <hip/hip_bf16.h>

typedef __attribute__((ext_vector_type(16))) _Float16 v16h;
typedef __attribute__((ext_vector_type(8)))  _Float16 v8h;
typedef __attribute__((ext_vector_type(8)))  float    v8f;

#define NEGV (-1e9f)

// sizes
#define NB    128      // batch
#define NPG   1024     // nodes per graph
#define DIM   128      // node feature dim
#define GDIM  128      // global feature dim
#define MAXC  512      // cubes per graph
#define MM    24       // moves
#define GG    8        // groups
#define NDIR  7
#define NAT   5
#define NPAIR 28

// ---------------------------------------------------------------------------
// Kernel 1: per-graph compaction of cube/group node indices (seg_pos replacement)
// ---------------------------------------------------------------------------
__global__ __launch_bounds__(64)
void index_kernel(const unsigned char* __restrict__ cube_mask,
                  const unsigned char* __restrict__ group_mask,
                  int* __restrict__ cube_node, int* __restrict__ group_node) {
  const int b = blockIdx.x;
  const int t = threadIdx.x;
  for (int i = t; i < MAXC; i += 64) cube_node[b * MAXC + i] = -1;
  if (t < GG) group_node[b * GG + t] = -1;
  __syncthreads();
  if (t == 0) {
    int cc = 0, gc = 0;
    const int base = b * NPG;
    for (int n = 0; n < NPG; ++n) {
      if (cube_mask[base + n])  { if (cc < MAXC) cube_node[b * MAXC + cc] = base + n; cc++; }
      if (group_mask[base + n]) { if (gc < GG)  group_node[b * GG  + gc] = base + n; gc++; }
    }
  }
}

// ---------------------------------------------------------------------------
// Kernel 2: CubeMoveHead — the big MLP via v_wmma_f32_16x16x32_f16.
// Grid 256 x 256 threads (8 waves). Each wave: two 16-row tiles (4096 total).
// ---------------------------------------------------------------------------
__global__ __launch_bounds__(256)
void cube_move_kernel(const float* __restrict__ nf, const float* __restrict__ gf,
                      const int* __restrict__ cube_node,
                      const unsigned char* __restrict__ move_mask,
                      const float* __restrict__ W1, const float* __restrict__ b1,
                      const float* __restrict__ W2, const float* __restrict__ b2,
                      float* __restrict__ out_cube) {
  extern __shared__ char smem[];
  _Float16* lds_w1 = (_Float16*)(smem);            // 32768 f16 (64 KB), B-swizzled 256x128
  _Float16* lds_w2 = (_Float16*)(smem + 65536);    //  4096 f16 (8 KB),  B-swizzled 128x32
  float*    lds_b1 = (float*)(smem + 73728);       //  128 f32
  float*    lds_b2 = (float*)(smem + 74240);       //   32 f32
  _Float16* lds_h  = (_Float16*)(smem + 74368);    // 8 waves * 16*128 f16 (32 KB)

  const int t = threadIdx.x;

  // Stage W1 into per-lane contiguous B-fragment layout:
  //   entry e = [kc(3)][nt(3)][lane(5)][i(4)] ; K = kc*32 + (lane>>4)*8 + (i&7) + (i>>3)*16 ; N = nt*16 + (lane&15)
  for (int e = t; e < 32768; e += 256) {
    const int i = e & 15, ln = (e >> 4) & 31, nt = (e >> 9) & 7, kc = e >> 12;
    const int k = kc * 32 + (ln >> 4) * 8 + (i & 7) + (i >> 3) * 16;
    const int n = nt * 16 + (ln & 15);
    lds_w1[e] = (_Float16)W1[k * 128 + n];
  }
  for (int e = t; e < 4096; e += 256) {
    const int i = e & 15, ln = (e >> 4) & 31, nt = (e >> 9) & 1, kc = e >> 10;
    const int k = kc * 32 + (ln >> 4) * 8 + (i & 7) + (i >> 3) * 16;
    const int n = nt * 16 + (ln & 15);
    lds_w2[e] = (n < MM) ? (_Float16)W2[k * MM + n] : (_Float16)0.0f;
  }
  if (t < 128) lds_b1[t] = b1[t];
  if (t < 32)  lds_b2[t] = (t < MM) ? b2[t] : 0.0f;
  __syncthreads();

  const int wave = t >> 5;
  const int lane = t & 31;
  const int m    = lane & 15;   // A row / C column index within tile
  const int hi   = lane >> 4;   // K-half selector for A, row-half for C
  _Float16* myh  = lds_h + wave * (16 * 128);

  for (int rep = 0; rep < 2; ++rep) {
    const int tile = blockIdx.x * 8 + wave + rep * 2048;  // 0..4095
    const int b    = tile >> 5;                            // 32 tiles per batch
    const int ct   = tile & 31;
    const int node = cube_node[b * MAXC + ct * 16 + m];    // this lane's A-row node
    const bool valid = node >= 0;
    const float* xn = nf + (size_t)(valid ? node : 0) * DIM;
    const float* xg = gf + (size_t)b * GDIM;

    v8f acc[8];
#pragma unroll
    for (int i = 0; i < 8; ++i) acc[i] = v8f{};

    // ---- layer 1: [16,256] x [256,128], K in 8 chunks of 32 ----
#pragma unroll
    for (int kc = 0; kc < 8; ++kc) {
      v16h a = {};
      if (valid) {
        const int K0 = kc * 32 + hi * 8;       // first contiguous 8
        const int K1 = K0 + 16;                // second contiguous 8
        const float* s0 = (kc < 4) ? (xn + K0) : (xg + (K0 - 128));
        const float* s1 = (kc < 4) ? (xn + K1) : (xg + (K1 - 128));
        const float4 p0 = *(const float4*)(s0);
        const float4 p1 = *(const float4*)(s0 + 4);
        const float4 q0 = *(const float4*)(s1);
        const float4 q1 = *(const float4*)(s1 + 4);
        a[0]=(_Float16)p0.x; a[1]=(_Float16)p0.y; a[2]=(_Float16)p0.z; a[3]=(_Float16)p0.w;
        a[4]=(_Float16)p1.x; a[5]=(_Float16)p1.y; a[6]=(_Float16)p1.z; a[7]=(_Float16)p1.w;
        a[8]=(_Float16)q0.x; a[9]=(_Float16)q0.y; a[10]=(_Float16)q0.z; a[11]=(_Float16)q0.w;
        a[12]=(_Float16)q1.x; a[13]=(_Float16)q1.y; a[14]=(_Float16)q1.z; a[15]=(_Float16)q1.w;
      }
#pragma unroll
      for (int nt = 0; nt < 8; ++nt) {
        const v16h bf = *(const v16h*)(lds_w1 + ((kc * 8 + nt) * 32 + lane) * 16);
        acc[nt] = __builtin_amdgcn_wmma_f32_16x16x32_f16(
            false, a, false, bf, (short)0, acc[nt], false, false);
      }
    }

    // ---- bias + relu, stage h (16x128 f16) in per-wave LDS ----
#pragma unroll
    for (int nt = 0; nt < 8; ++nt) {
      const int ncol = nt * 16 + m;
      const float bia = lds_b1[ncol];
#pragma unroll
      for (int v = 0; v < 8; ++v) {
        const int row = v + hi * 8;
        myh[row * 128 + ncol] = (_Float16)fmaxf(acc[nt][v] + bia, 0.0f);
      }
    }

    // ---- layer 2: [16,128] x [128,32] (cols 24..31 are zero pad) ----
    v8f a2c[2]; a2c[0] = v8f{}; a2c[1] = v8f{};
#pragma unroll
    for (int kc = 0; kc < 4; ++kc) {
      const int K0 = kc * 32 + hi * 8;
      const v8h lo = *(const v8h*)(myh + m * 128 + K0);
      const v8h hh = *(const v8h*)(myh + m * 128 + K0 + 16);
      v16h a;
#pragma unroll
      for (int i = 0; i < 8; ++i) { a[i] = lo[i]; a[8 + i] = hh[i]; }
#pragma unroll
      for (int nt = 0; nt < 2; ++nt) {
        const v16h bf = *(const v16h*)(lds_w2 + ((kc * 2 + nt) * 32 + lane) * 16);
        a2c[nt] = __builtin_amdgcn_wmma_f32_16x16x32_f16(
            false, a, false, bf, (short)0, a2c[nt], false, false);
      }
    }

    // ---- scatter to cube_logits with validity + move_mask ----
#pragma unroll
    for (int v = 0; v < 8; ++v) {
      const int row = v + hi * 8;
      const int cc  = ct * 16 + row;
      const bool rowok = cube_node[b * MAXC + cc] >= 0;
      const size_t rbase = ((size_t)b * MAXC + cc) * MM;
#pragma unroll
      for (int nt = 0; nt < 2; ++nt) {
        const int ncol = nt * 16 + m;
        if (ncol < MM) {
          const float val = a2c[nt][v] + lds_b2[ncol];
          out_cube[rbase + ncol] = (rowok && move_mask[rbase + ncol]) ? val : NEGV;
        }
      }
    }
  }
}

// ---------------------------------------------------------------------------
// Kernel 3: ActionTypeHead (tiny MLP, VALU)
// ---------------------------------------------------------------------------
__global__ __launch_bounds__(64)
void at_kernel(const float* __restrict__ gf, const unsigned char* __restrict__ mask,
               const float* __restrict__ W1, const float* __restrict__ b1,
               const float* __restrict__ W2, const float* __restrict__ b2,
               float* __restrict__ out) {
  __shared__ float gsh[128];
  __shared__ float h[64];
  const int b = blockIdx.x, t = threadIdx.x;
  gsh[t] = gf[b * GDIM + t];
  gsh[t + 64] = gf[b * GDIM + t + 64];
  __syncthreads();
  float acc = b1[t];
  for (int k = 0; k < 128; ++k) acc += gsh[k] * W1[k * 64 + t];
  h[t] = fmaxf(acc, 0.0f);
  __syncthreads();
  if (t < NAT) {
    float o = b2[t];
    for (int k = 0; k < 64; ++k) o += h[k] * W2[k * NAT + t];
    out[b * NAT + t] = mask[b * NAT + t] ? o : NEGV;
  }
}

// ---------------------------------------------------------------------------
// Kernel 4: Docking (28 pairs MLP3) + Maneuver (8 groups MLP2), per-batch block
// ---------------------------------------------------------------------------
__device__ const int D_IU[NPAIR] = {0,0,0,0,0,0,0,1,1,1,1,1,1,2,2,2,2,2,3,3,3,3,4,4,4,5,5,6};
__device__ const int D_JU[NPAIR] = {1,2,3,4,5,6,7,2,3,4,5,6,7,3,4,5,6,7,4,5,6,7,5,6,7,6,7,7};

__global__ __launch_bounds__(128)
void dock_man_kernel(const float* __restrict__ nf, const float* __restrict__ gf,
                     const int* __restrict__ group_node,
                     const unsigned char* __restrict__ dock_mask,
                     const unsigned char* __restrict__ man_mask,
                     const float* __restrict__ dkW1, const float* __restrict__ dkb1,
                     const float* __restrict__ dkW2, const float* __restrict__ dkb2,
                     const float* __restrict__ dkW3, const float* __restrict__ dkb3,
                     const float* __restrict__ mnW1, const float* __restrict__ mnb1,
                     const float* __restrict__ mnW2, const float* __restrict__ mnb2,
                     float* __restrict__ out_dock, float* __restrict__ out_man) {
  __shared__ float gfeat[GG * DIM];
  __shared__ float gsh[GDIM];
  __shared__ float h1[128];
  __shared__ float h2[64];
  __shared__ int hasg[GG];
  const int b = blockIdx.x, t = threadIdx.x;

  gsh[t] = gf[b * GDIM + t];
  if (t < GG) hasg[t] = (group_node[b * GG + t] >= 0);
  for (int idx = t; idx < GG * DIM; idx += 128) {
    const int g = idx >> 7, k = idx & 127;
    const int node = group_node[b * GG + g];
    gfeat[idx] = (node >= 0) ? nf[(size_t)node * DIM + k] : 0.0f;
  }
  __syncthreads();

  // --- DockingHead: MLP3 384 -> 128 -> 64 -> 1 over 28 pairs ---
  for (int p = 0; p < NPAIR; ++p) {
    const int i = D_IU[p], j = D_JU[p];
    const float* fi = &gfeat[i * DIM];
    const float* fj = &gfeat[j * DIM];
    float acc = dkb1[t];
    for (int k = 0; k < 128; ++k) acc += fi[k] * dkW1[k * 128 + t];
    for (int k = 0; k < 128; ++k) acc += fj[k] * dkW1[(128 + k) * 128 + t];
    for (int k = 0; k < 128; ++k) acc += gsh[k] * dkW1[(256 + k) * 128 + t];
    h1[t] = fmaxf(acc, 0.0f);
    __syncthreads();
    if (t < 64) {
      float a2 = dkb2[t];
      for (int k = 0; k < 128; ++k) a2 += h1[k] * dkW2[k * 64 + t];
      h2[t] = fmaxf(a2, 0.0f);
    }
    __syncthreads();
    if (t == 0) {
      float s = dkb3[0];
      for (int k = 0; k < 64; ++k) s += h2[k] * dkW3[k];
      const bool ok = dock_mask[b * NPAIR + p] && hasg[i] && hasg[j];
      out_dock[b * NPAIR + p] = ok ? s : NEGV;
    }
    __syncthreads();
  }

  // --- ManeuverHead: MLP2 256 -> 128 -> 7 over 8 groups ---
  for (int g = 0; g < GG; ++g) {
    const float* fg = &gfeat[g * DIM];
    float acc = mnb1[t];
    for (int k = 0; k < 128; ++k) acc += fg[k] * mnW1[k * 128 + t];
    for (int k = 0; k < 128; ++k) acc += gsh[k] * mnW1[(128 + k) * 128 + t];
    h1[t] = fmaxf(acc, 0.0f);
    __syncthreads();
    if (t < NDIR) {
      float o = mnb2[t];
      for (int k = 0; k < 128; ++k) o += h1[k] * mnW2[k * NDIR + t];
      o = hasg[g] ? o : NEGV;
      const int oi = b * (GG * NDIR) + g * NDIR + t;
      out_man[oi] = man_mask[oi] ? o : NEGV;
    }
    __syncthreads();
  }
}

// ---------------------------------------------------------------------------
extern "C" void kernel_launch(void* const* d_in, const int* in_sizes, int n_in,
                              void* d_out, int out_size, void* d_ws, size_t ws_size,
                              hipStream_t stream) {
  (void)in_sizes; (void)n_in; (void)out_size; (void)ws_size;
  const float* nf = (const float*)d_in[0];
  const float* gf = (const float*)d_in[1];
  const unsigned char* at_mask    = (const unsigned char*)d_in[2];
  const unsigned char* cube_mask  = (const unsigned char*)d_in[3];
  const unsigned char* group_mask = (const unsigned char*)d_in[4];
  /* d_in[5] = batch (int32) — layout is implicit (contiguous segments) */
  const unsigned char* move_mask  = (const unsigned char*)d_in[6];
  const unsigned char* dock_mask  = (const unsigned char*)d_in[7];
  const unsigned char* man_mask   = (const unsigned char*)d_in[8];
  const float* at_W1 = (const float*)d_in[9];
  const float* at_b1 = (const float*)d_in[10];
  const float* at_W2 = (const float*)d_in[11];
  const float* at_b2 = (const float*)d_in[12];
  const float* cm_W1 = (const float*)d_in[13];
  const float* cm_b1 = (const float*)d_in[14];
  const float* cm_W2 = (const float*)d_in[15];
  const float* cm_b2 = (const float*)d_in[16];
  const float* dk_W1 = (const float*)d_in[17];
  const float* dk_b1 = (const float*)d_in[18];
  const float* dk_W2 = (const float*)d_in[19];
  const float* dk_b2 = (const float*)d_in[20];
  const float* dk_W3 = (const float*)d_in[21];
  const float* dk_b3 = (const float*)d_in[22];
  const float* mn_W1 = (const float*)d_in[23];
  const float* mn_b1 = (const float*)d_in[24];
  const float* mn_W2 = (const float*)d_in[25];
  const float* mn_b2 = (const float*)d_in[26];

  float* out      = (float*)d_out;
  float* out_at   = out;                                  // [128,5]
  float* out_cube = out + NB * NAT;                       // [128,512*24]
  float* out_dock = out_cube + (size_t)NB * MAXC * MM;    // [128,28]
  float* out_man  = out_dock + NB * NPAIR;                // [128,56]

  int* cube_node  = (int*)d_ws;                           // 128*512 ints
  int* group_node = (int*)((char*)d_ws + (size_t)NB * MAXC * sizeof(int));

  index_kernel<<<NB, 64, 0, stream>>>(cube_mask, group_mask, cube_node, group_node);

  const size_t smem = 65536 + 8192 + 512 + 128 + 8 * 16 * 128 * sizeof(_Float16); // 107136 B
  cube_move_kernel<<<256, 256, smem, stream>>>(nf, gf, cube_node, move_mask,
                                               cm_W1, cm_b1, cm_W2, cm_b2, out_cube);

  at_kernel<<<NB, 64, 0, stream>>>(gf, at_mask, at_W1, at_b1, at_W2, at_b2, out_at);

  dock_man_kernel<<<NB, 128, 0, stream>>>(nf, gf, group_node, dock_mask, man_mask,
                                          dk_W1, dk_b1, dk_W2, dk_b2, dk_W3, dk_b3,
                                          mn_W1, mn_b1, mn_W2, mn_b2, out_dock, out_man);
}
